// Attention_6554120094055
// MI455X (gfx1250) — matrix operand order repaired and checked by reading, not compiled, not run on hardware
//
#include <hip/hip_runtime.h>

#define N_HEADS 16
#define N_KV_HEADS 4
#define HEAD_DIM 128

typedef __attribute__((ext_vector_type(16))) __bf16 v16bf;
typedef __attribute__((ext_vector_type(8)))  float  v8f;
typedef __attribute__((ext_vector_type(4)))  float  v4f;
typedef __attribute__((ext_vector_type(4)))  unsigned int v4u;
typedef __attribute__((ext_vector_type(2)))  unsigned int v2u;
typedef __attribute__((ext_vector_type(4)))  unsigned int u32x4;
typedef __attribute__((ext_vector_type(8)))  int i32x8;
typedef __attribute__((ext_vector_type(4)))  int i32x4;

union FragU { v16bf v; v4u u[2]; };

__device__ __forceinline__ unsigned short f2bf(float x) {
  unsigned int u = __builtin_bit_cast(unsigned int, x);
  u += 0x7FFFu + ((u >> 16) & 1u);            // round-to-nearest-even
  return (unsigned short)(u >> 16);
}
__device__ __forceinline__ float bf2f(unsigned short u) {
  return __builtin_bit_cast(float, ((unsigned int)u) << 16);
}
__device__ __forceinline__ v2u pack4(v4f f) {
  v2u r;
  r.x = (unsigned int)f2bf(f.x) | ((unsigned int)f2bf(f.y) << 16);
  r.y = (unsigned int)f2bf(f.z) | ((unsigned int)f2bf(f.w) << 16);
  return r;
}
__device__ __forceinline__ unsigned lds_off(const void* p) {
  // generic pointers to LDS carry the LDS byte offset in the low 32 bits
  return (unsigned)(unsigned long long)(uintptr_t)p;
}

// ---- Tensor Data Mover: 2D bf16 tile (tileK x tileRows) -> LDS (contiguous rows) ----
// D# per CDNA5 ISA ch.8: group0 {count, lds_addr, global_addr, type=2},
// group1 {data_size=2B, tensor_dim0/1, tile_dim0/1, tensor_dim0_stride}.
__device__ __forceinline__ void tdm_load_2d(const unsigned short* gptr, unsigned ldsoff,
                                            unsigned tileK, unsigned tileRows,
                                            unsigned strideElems) {
  unsigned long long ga = (unsigned long long)(uintptr_t)gptr;
  u32x4 g0;
  g0.x = 1u;                                             // count=1, user descriptor
  g0.y = ldsoff;                                         // lds_addr (bytes)
  g0.z = (unsigned)ga;                                   // global_addr[31:0]
  g0.w = (unsigned)((ga >> 32) & 0x01ffffffu) | (2u << 30);  // addr[56:32] | type=2
  const unsigned td0 = strideElems;                      // tensor_dim0 (row length)
  const unsigned td1 = 0x00100000u;                      // tensor_dim1 (large, in-bounds)
  i32x8 g1;
  g1[0] = 0x00010000;                                    // data_size=1 -> 2 bytes
  g1[1] = (int)((td0 & 0xffffu) << 16);                  // tensor_dim0[15:0]
  g1[2] = (int)(((td0 >> 16) & 0xffffu) | ((td1 & 0xffffu) << 16));
  g1[3] = (int)(((td1 >> 16) & 0xffffu) | (tileK << 16)); // tile_dim0
  g1[4] = (int)(tileRows & 0xffffu);                     // tile_dim1 (tile_dim2=0)
  g1[5] = (int)strideElems;                              // tensor_dim0_stride[31:0]
  g1[6] = 0;
  g1[7] = 0;
  i32x4 z4 = {0, 0, 0, 0};                               // 2D: groups 2/3 unused
  i32x8 z8 = {0, 0, 0, 0, 0, 0, 0, 0};
  __builtin_amdgcn_tensor_load_to_lds(g0, g1, z4, z4, z8, 0);
}

// A fragment (16x32 bf16, M x K): element (r,k) at base[r*ld + k].
__device__ __forceinline__ v16bf frag_A(const unsigned short* base, int ld,
                                        int row_base, int kbase, int lane) {
  int r = row_base + (lane & 15);
  int h = (lane >> 4) & 1;
  const unsigned short* p = base + r * ld + kbase;
  FragU f;
  f.u[0] = *(const v4u*)(p + h * 8);
  f.u[1] = *(const v4u*)(p + 16 + h * 8);
  return f.v;
}
// B fragment (32x16 bf16, K x N): element (k,c) at base[c*ld + k].
__device__ __forceinline__ v16bf frag_B(const unsigned short* base, int ld,
                                        int col_base, int kbase, int lane) {
  int c = col_base + (lane & 15);
  int h = (lane >> 4) & 1;
  const unsigned short* p = base + c * ld + kbase;
  FragU f;
  f.u[0] = *(const v4u*)(p + h * 8);
  f.u[1] = *(const v4u*)(p + 16 + h * 8);
  return f.v;
}

// ---- f32 -> bf16 bulk conversion ----
__global__ __launch_bounds__(256) void f32_to_bf16(const float* __restrict__ src,
                                                   unsigned short* __restrict__ dst,
                                                   int total4) {
  int i = blockIdx.x * blockDim.x + threadIdx.x;
  if (i >= total4) return;
  v4f f = *(const v4f*)(src + (size_t)i * 4);
  *(v2u*)(dst + (size_t)i * 4) = pack4(f);
}

#define BM 128
#define BN 64
#define BK 64

// C[m][n] = sum_k A[m][k]*W[n][k]; A: MxK bf16, W: NxK bf16; C f32 or bf16.
// TDM-staged, double-buffered LDS. 8 wave32, wave tile 32x32.
__global__ __launch_bounds__(256) void gemm_bf16_tdm(const unsigned short* __restrict__ A,
                                                     const unsigned short* __restrict__ W,
                                                     void* __restrict__ Cout,
                                                     int M, int N, int K, int out_bf16) {
  __shared__ unsigned short As[2][BM][BK];   // 2 x 16 KB
  __shared__ unsigned short Bs[2][BN][BK];   // 2 x  8 KB
  const int tid  = threadIdx.x;
  const int lane = tid & 31;
  const int wave = tid >> 5;
  const int bm = blockIdx.y * BM;
  const int bn = blockIdx.x * BN;
  const int wm = (wave >> 1) * 32;
  const int wn = (wave & 1) * 32;
  v8f acc[2][2] = {};

  if (wave == 0) {
    tdm_load_2d(A + (size_t)bm * K, lds_off(&As[0][0][0]), BK, BM, (unsigned)K);
    tdm_load_2d(W + (size_t)bn * K, lds_off(&Bs[0][0][0]), BK, BN, (unsigned)K);
  }
  int buf = 0;
  for (int k0 = 0; k0 < K; k0 += BK) {
    if (wave == 0) __builtin_amdgcn_s_wait_tensorcnt(0);  // current buf DMA done
    __syncthreads();                                      // publish buf / protect buf^1
    if (wave == 0 && k0 + BK < K) {                       // prefetch next stage
      tdm_load_2d(A + (size_t)bm * K + (k0 + BK), lds_off(&As[buf ^ 1][0][0]), BK, BM, (unsigned)K);
      tdm_load_2d(W + (size_t)bn * K + (k0 + BK), lds_off(&Bs[buf ^ 1][0][0]), BK, BN, (unsigned)K);
    }
#pragma unroll
    for (int kk = 0; kk < BK; kk += 32) {
      v16bf ra0 = frag_A(&As[buf][0][0], BK, wm,      kk, lane);
      v16bf ra1 = frag_A(&As[buf][0][0], BK, wm + 16, kk, lane);
      v16bf rb0 = frag_B(&Bs[buf][0][0], BK, wn,      kk, lane);
      v16bf rb1 = frag_B(&Bs[buf][0][0], BK, wn + 16, kk, lane);
      acc[0][0] = __builtin_amdgcn_wmma_f32_16x16x32_bf16(false, ra0, false, rb0, (short)0, acc[0][0], false, false);
      acc[0][1] = __builtin_amdgcn_wmma_f32_16x16x32_bf16(false, ra0, false, rb1, (short)0, acc[0][1], false, false);
      acc[1][0] = __builtin_amdgcn_wmma_f32_16x16x32_bf16(false, ra1, false, rb0, (short)0, acc[1][0], false, false);
      acc[1][1] = __builtin_amdgcn_wmma_f32_16x16x32_bf16(false, ra1, false, rb1, (short)0, acc[1][1], false, false);
    }
    buf ^= 1;
  }

  const int l16 = lane & 15;
  const int h2  = (lane >> 4) & 1;
#pragma unroll
  for (int i = 0; i < 2; ++i)
#pragma unroll
    for (int j = 0; j < 2; ++j) {
      int col  = bn + wn + 16 * j + l16;
      int rowb = bm + wm + 16 * i + 8 * h2;
      if (out_bf16) {
        unsigned short* Cb = (unsigned short*)Cout;
#pragma unroll
        for (int v = 0; v < 8; ++v)
          Cb[(size_t)(rowb + v) * N + col] = f2bf(acc[i][j][v]);
      } else {
        float* Cf = (float*)Cout;
#pragma unroll
        for (int v = 0; v < 8; ++v)
          Cf[(size_t)(rowb + v) * N + col] = acc[i][j][v];
      }
    }
}

// In-place interleaved RoPE on bf16 [M][H*128]; one thread per (pos,head,pair).
__global__ __launch_bounds__(256) void rope_bf16(unsigned short* __restrict__ T,
                                                 int H, int S, int total) {
  int idx = blockIdx.x * blockDim.x + threadIdx.x;
  if (idx >= total) return;
  int i    = idx & 63;
  int rest = idx >> 6;
  int hh   = rest % H;
  int m    = rest / H;
  int s    = m % S;
  float inv = __expf(-0.14391156831212787f * (float)i);   // 10000^(-2i/128)
  float ang = (float)s * inv;
  float c, sn;
  __sincosf(ang, &sn, &c);
  size_t off = (size_t)m * (H * HEAD_DIM) + hh * HEAD_DIM + 2 * i;
  float t0 = bf2f(T[off]), t1 = bf2f(T[off + 1]);
  T[off]     = f2bf(t0 * c - t1 * sn);
  T[off + 1] = f2bf(t0 * sn + t1 * c);
}

// Causal flash attention, bf16 in / bf16 out. Block = (b,h,64 q rows); 4 wave32.
// Q and K tiles staged by TDM; V staged transposed manually.
__global__ __launch_bounds__(128) void flash_attn(const unsigned short* __restrict__ Q,
                                                  const unsigned short* __restrict__ Kg,
                                                  const unsigned short* __restrict__ Vg,
                                                  unsigned short* __restrict__ Og,
                                                  int S) {
  __shared__ unsigned short Qs[64][HEAD_DIM];   // 16 KB (TDM)
  __shared__ unsigned short Ks[32][HEAD_DIM];   //  8 KB (TDM) [key][hd]
  __shared__ unsigned short Vt[HEAD_DIM][32];   //  8 KB [hd][key]
  __shared__ unsigned short Ps[4][16][32];      //  4 KB per-wave P tile
  const int tid  = threadIdx.x;
  const int lane = tid & 31;
  const int wave = tid >> 5;
  const int b    = blockIdx.x / N_HEADS;
  const int h    = blockIdx.x % N_HEADS;
  const int kvh  = h / (N_HEADS / N_KV_HEADS);
  const int qb   = blockIdx.y * 64;
  const int qr   = wave * 16;
  const int l16  = lane & 15;
  const int h2   = (lane >> 4) & 1;

  if (wave == 0)
    tdm_load_2d(Q + (size_t)(b * S + qb) * (N_HEADS * HEAD_DIM) + h * HEAD_DIM,
                lds_off(&Qs[0][0]), HEAD_DIM, 64, N_HEADS * HEAD_DIM);

  v8f o[8] = {};
  float mstate[8], lsum[8];
#pragma unroll
  for (int v = 0; v < 8; ++v) { mstate[v] = -__builtin_inff(); lsum[v] = 0.f; }

  const int nchunks = (qb + 64) / 32;
  const float scale = 0.08838834764831845f;     // 1/sqrt(128)

  for (int kc = 0; kc < nchunks; ++kc) {
    const int kbase = kc * 32;
    __syncthreads();                            // previous chunk's readers done
    if (wave == 0)
      tdm_load_2d(Kg + (size_t)(b * S + kbase) * (N_KV_HEADS * HEAD_DIM) + kvh * HEAD_DIM,
                  lds_off(&Ks[0][0]), HEAD_DIM, 32, N_KV_HEADS * HEAD_DIM);
#pragma unroll
    for (int it = 0; it < 8; ++it) {            // V: 32*128 elems, 4 per thread/iter
      int i  = tid + it * 128;
      int kk = i >> 5;
      int d  = (i & 31) * 4;
      v2u r = *(const v2u*)(Vg + (size_t)(b * S + kbase + kk) * (N_KV_HEADS * HEAD_DIM)
                                 + kvh * HEAD_DIM + d);
      Vt[d + 0][kk] = (unsigned short)(r.x & 0xffffu);
      Vt[d + 1][kk] = (unsigned short)(r.x >> 16);
      Vt[d + 2][kk] = (unsigned short)(r.y & 0xffffu);
      Vt[d + 3][kk] = (unsigned short)(r.y >> 16);
    }
    if (wave == 0) __builtin_amdgcn_s_wait_tensorcnt(0);  // Q(first pass) + K ready
    __syncthreads();

    // S = Q * K^T  (16 x 32 per wave)
    v8f sc[2] = {};
#pragma unroll
    for (int c = 0; c < 4; ++c) {
      v16bf aq  = frag_A(&Qs[0][0], HEAD_DIM, qr, c * 32, lane);
      v16bf bk0 = frag_B(&Ks[0][0], HEAD_DIM, 0,  c * 32, lane);
      v16bf bk1 = frag_B(&Ks[0][0], HEAD_DIM, 16, c * 32, lane);
      sc[0] = __builtin_amdgcn_wmma_f32_16x16x32_bf16(false, aq, false, bk0, (short)0, sc[0], false, false);
      sc[1] = __builtin_amdgcn_wmma_f32_16x16x32_bf16(false, aq, false, bk1, (short)0, sc[1], false, false);
    }

    // online softmax in C layout
#pragma unroll
    for (int v = 0; v < 8; ++v) {
      int row = qb + qr + v + 8 * h2;
      float s0 = sc[0][v] * scale;
      float s1 = sc[1][v] * scale;
      if (kbase + l16 > row)      s0 = -1e30f;
      if (kbase + 16 + l16 > row) s1 = -1e30f;
      float rm = fmaxf(s0, s1);
#pragma unroll
      for (int off = 1; off < 16; off <<= 1) rm = fmaxf(rm, __shfl_xor(rm, off, 32));
      float mo = mstate[v];
      float mn = fmaxf(mo, rm);
      float alpha = __expf(mo - mn);
      float p0 = __expf(s0 - mn);
      float p1 = __expf(s1 - mn);
      float rs = p0 + p1;
#pragma unroll
      for (int off = 1; off < 16; off <<= 1) rs += __shfl_xor(rs, off, 32);
      lsum[v]   = lsum[v] * alpha + rs;
      mstate[v] = mn;
#pragma unroll
      for (int t = 0; t < 8; ++t) o[t][v] *= alpha;
      int prow = v + 8 * h2;
      Ps[wave][prow][l16]      = f2bf(p0);
      Ps[wave][prow][16 + l16] = f2bf(p1);
    }

    asm volatile("s_wait_dscnt 0x0" ::: "memory");  // wave-private LDS round trip

    // O += P * V
    v16bf ap = frag_A(&Ps[wave][0][0], 32, 0, 0, lane);
#pragma unroll
    for (int t = 0; t < 8; ++t) {
      v16bf bv = frag_B(&Vt[0][0], 32, t * 16, 0, lane);
      o[t] = __builtin_amdgcn_wmma_f32_16x16x32_bf16(false, ap, false, bv, (short)0, o[t], false, false);
    }
  }

#pragma unroll
  for (int v = 0; v < 8; ++v) {
    int row = qb + qr + v + 8 * h2;
    float inv = 1.0f / lsum[v];
#pragma unroll
    for (int t = 0; t < 8; ++t)
      Og[(size_t)(b * S + row) * (N_HEADS * HEAD_DIM) + h * HEAD_DIM + t * 16 + l16] =
          f2bf(o[t][v] * inv);
  }
}

extern "C" void kernel_launch(void* const* d_in, const int* in_sizes, int n_in,
                              void* d_out, int out_size, void* d_ws, size_t ws_size,
                              hipStream_t stream) {
  (void)in_sizes; (void)n_in; (void)out_size; (void)ws_size;
  const float* x  = (const float*)d_in[0];
  const float* wq = (const float*)d_in[1];
  const float* wk = (const float*)d_in[2];
  const float* wv = (const float*)d_in[3];
  const float* wo = (const float*)d_in[4];

  const int Bb = 2, S = 2048, D = 2048;
  const int M = Bb * S;                     // 4096

  unsigned short* xb  = (unsigned short*)d_ws;                 // M x 2048
  unsigned short* wqb = xb  + (size_t)M * D;                   // 2048 x 2048
  unsigned short* wkb = wqb + (size_t)2048 * D;                // 512 x 2048
  unsigned short* wvb = wkb + (size_t)512 * D;                 // 512 x 2048
  unsigned short* wob = wvb + (size_t)512 * D;                 // 2048 x 2048
  unsigned short* qb  = wob + (size_t)2048 * D;                // M x 2048
  unsigned short* kb  = qb  + (size_t)M * 2048;                // M x 512
  unsigned short* vb  = kb  + (size_t)M * 512;                 // M x 512
  unsigned short* ab  = vb  + (size_t)M * 512;                 // M x 2048

  dim3 blk(256);
  f32_to_bf16<<<(M * D / 4 + 255) / 256, blk, 0, stream>>>(x, xb, M * D / 4);
  f32_to_bf16<<<(2048 * D / 4 + 255) / 256, blk, 0, stream>>>(wq, wqb, 2048 * D / 4);
  f32_to_bf16<<<(512 * D / 4 + 255) / 256, blk, 0, stream>>>(wk, wkb, 512 * D / 4);
  f32_to_bf16<<<(512 * D / 4 + 255) / 256, blk, 0, stream>>>(wv, wvb, 512 * D / 4);
  f32_to_bf16<<<(2048 * D / 4 + 255) / 256, blk, 0, stream>>>(wo, wob, 2048 * D / 4);

  gemm_bf16_tdm<<<dim3(2048 / BN, M / BM), blk, 0, stream>>>(xb, wqb, qb, M, 2048, D, 1);
  gemm_bf16_tdm<<<dim3(512 / BN, M / BM), blk, 0, stream>>>(xb, wkb, kb, M, 512, D, 1);
  gemm_bf16_tdm<<<dim3(512 / BN, M / BM), blk, 0, stream>>>(xb, wvb, vb, M, 512, D, 1);

  int totq = M * N_HEADS * (HEAD_DIM / 2);
  rope_bf16<<<(totq + 255) / 256, blk, 0, stream>>>(qb, N_HEADS, S, totq);
  int totk = M * N_KV_HEADS * (HEAD_DIM / 2);
  rope_bf16<<<(totk + 255) / 256, blk, 0, stream>>>(kb, N_KV_HEADS, S, totk);

  flash_attn<<<dim3(Bb * N_HEADS, S / 64), dim3(128), 0, stream>>>(qb, kb, vb, ab, S);

  gemm_bf16_tdm<<<dim3(D / BN, M / BM), blk, 0, stream>>>(ab, wob, d_out, M, D, D, 0);
}
